// HistogramLoss_66735201845351
// MI455X (gfx1250) — matrix operand order repaired and checked by reading, not compiled, not run on hardware
//
#include <hip/hip_runtime.h>
#include <hip/hip_bf16.h>

// HistogramLoss on MI455X (gfx1250):
//   F (N=4096, D=512, fp32, rows L2-normalized) -> pairwise S = F F^T (upper
//   triangle), triangular soft-binning into 151 bins split by class equality,
//   loss = sum(hist_neg * cumsum(hist_pos)).
// Roofline: 2*N^2*D ~= 17.2 GFLOP over 8 MB of input -> compute bound; use
// v_wmma_f32_16x16x32_f16 (f16 in, f32 acc). f16 rounding error (~1e-4 on s)
// is far below the bin width (1/75), so the matrix-core path is safe.
// This revision: 64x64 wave tiles (4x4 WMMA accs) -> 16 b128 loads per
// 16 WMMAs (was 12 per 8), cutting vector-memory traffic per FLOP by 33%.

#define NUM_STEPS 151
#define HALF_BINS 75          // (NUM_STEPS-1)/2 ; 1/STEP = 75
#define HIST_PAD  152         // 151 bins + 1 pad slot
#define HIST2     (2 * HIST_PAD)  // [0,152): pos ; [152,304): neg
#define NUM_CLASSES 100
#define TILE_M 128            // block rows  (2 waves x 64)
#define TILE_N 256            // block cols  (4 waves x 64)
#define WAVES_PER_BLOCK 8

typedef __attribute__((ext_vector_type(8)))  _Float16 v8h;
typedef __attribute__((ext_vector_type(16))) _Float16 v16h;
typedef __attribute__((ext_vector_type(8)))  float    v8f;

// --- fragment loaders -------------------------------------------------------
// A (16x32, MxK) per ISA table: lane half h holds, for row M=lane&15,
//   VGPR0-3: K = k0 + 8h + {0..7};  VGPR4-7: K = k0 + 16 + 8h + {0..7}
__device__ __forceinline__ v16h load_a_frag(const _Float16* __restrict__ row,
                                            int k0, int h) {
  v8h lo = *(const v8h*)(row + k0 + 8 * h);
  v8h hi = *(const v8h*)(row + k0 + 8 * h + 16);
  return __builtin_shufflevector(lo, hi, 0, 1, 2, 3, 4, 5, 6, 7,
                                 8, 9, 10, 11, 12, 13, 14, 15);
}
// B (32x16, KxN) column-major from F rows (B = F^T): lane half h holds, for
// col N=lane&15, VGPR0-3: K = k0 + 8h + {0..7};  VGPR4-7: K = k0 + 16 + 8h + {0..7}
__device__ __forceinline__ v16h load_b_frag(const _Float16* __restrict__ row,
                                            int k0, int h) {
  v8h lo = *(const v8h*)(row + k0 + 8 * h);
  v8h hi = *(const v8h*)(row + k0 + 8 * h + 16);
  return __builtin_shufflevector(lo, hi, 0, 1, 2, 3, 4, 5, 6, 7,
                                 8, 9, 10, 11, 12, 13, 14, 15);
}

#define WMMA_F16(A, B, C)                                                      \
  __builtin_amdgcn_wmma_f32_16x16x32_f16(false, (A), false, (B), (short)0,     \
                                         (C), false, false)

// --- kernel 0: zero the global histogram accumulator ------------------------
__global__ void zero_hist_kernel(float* __restrict__ gHist) {
  int i = threadIdx.x;
  if (i < HIST2) gHist[i] = 0.0f;
}

// --- kernel 1: fp32 -> fp16 features ----------------------------------------
__global__ void cvt_f16_kernel(const float* __restrict__ x,
                               _Float16* __restrict__ y, int n) {
  int i = blockIdx.x * blockDim.x + threadIdx.x;
  if (i < n) y[i] = (_Float16)x[i];
}

// --- kernel 2: fused WMMA GEMM + soft-binned histogram ----------------------
// Grid (n/TILE_N, n/TILE_M), block 256 = 8 waves (2 along M x 4 along N).
// Wave tile 64x64 = 4x4 WMMA 16x16 accumulators.
__global__ __launch_bounds__(256)
void hist_gemm_kernel(const _Float16* __restrict__ Fh,
                      const int* __restrict__ classes,
                      float* __restrict__ gHist, int d) {
  const int bit = blockIdx.y;   // row tile (TILE_M rows)
  const int bjt = blockIdx.x;   // col tile (TILE_N cols)
  // Keep block iff it intersects the strict upper triangle:
  // exists i<j with i in [bit*128,+128), j in [bjt*256,+256)  <=>  bjt >= bit/2
  if (bjt < (bit >> 1)) return;

  __shared__ float sH[WAVES_PER_BLOCK * HIST2];  // per-wave histograms
  __shared__ int clsRow[TILE_M];
  __shared__ int clsCol[TILE_N];

  const int tid = threadIdx.x;
  for (int b = tid; b < WAVES_PER_BLOCK * HIST2; b += 256) sH[b] = 0.0f;
  if (tid < TILE_M) clsRow[tid] = classes[bit * TILE_M + tid];
  for (int i = tid; i < TILE_N; i += 256) clsCol[i] = classes[bjt * TILE_N + i];
  __syncthreads();

  const int lane = tid & 31;
  const int wave = tid >> 5;
  const int wm = wave >> 2;   // 2 waves along M -> 64 rows each
  const int wn = wave & 3;    // 4 waves along N -> 64 cols each
  const int h = lane >> 4;    // lane half
  const int r16 = lane & 15;

  const int rowBase = bit * TILE_M + wm * 64;
  const int colBase = bjt * TILE_N + wn * 64;

  const _Float16* arow[4];
  const _Float16* brow[4];
#pragma unroll
  for (int t = 0; t < 4; ++t) {
    arow[t] = Fh + (size_t)(rowBase + 16 * t + r16) * d;
    brow[t] = Fh + (size_t)(colBase + 16 * t + r16) * d;
  }

  v8f acc[4][4];
#pragma unroll
  for (int mt = 0; mt < 4; ++mt)
#pragma unroll
    for (int nt = 0; nt < 4; ++nt) acc[mt][nt] = (v8f){};

  for (int k0 = 0; k0 < d; k0 += 32) {
    v16h a[4], b[4];
#pragma unroll
    for (int t = 0; t < 4; ++t) {
      a[t] = load_a_frag(arow[t], k0, h);
      b[t] = load_b_frag(brow[t], k0, h);
    }
#pragma unroll
    for (int mt = 0; mt < 4; ++mt)
#pragma unroll
      for (int nt = 0; nt < 4; ++nt)
        acc[mt][nt] = WMMA_F16(a[mt], b[nt], acc[mt][nt]);
  }

  // Epilogue: soft-bin each C element into this wave's private LDS histogram.
  // C/D layout: VGPR e, lane -> M = e + 8*(lane>>4), N = lane&15.
  float* myH = &sH[wave * HIST2];
#pragma unroll
  for (int mt = 0; mt < 4; ++mt) {
#pragma unroll
    for (int nt = 0; nt < 4; ++nt) {
      const int lc = wn * 64 + 16 * nt + r16;  // local col in block tile
      const int gj = bjt * TILE_N + lc;
      const int cj = clsCol[lc];
#pragma unroll
      for (int e = 0; e < 8; ++e) {
        const int lr = wm * 64 + 16 * mt + 8 * h + e;  // local row
        const int gi = bit * TILE_M + lr;
        if (gi >= gj) continue;                        // strict upper triangle
        const float s = acc[mt][nt][e];
        const float t = s * (float)HALF_BINS;          // s / STEP
        const float fl = floorf(t);
        const int jb = (int)fl + HALF_BINS;
        const float wa = t - fl;
        const float wb = 1.0f - wa;
        const int base = (clsRow[lr] == cj) ? 0 : HIST_PAD;
        if ((unsigned)jb < (unsigned)NUM_STEPS) atomicAdd(&myH[base + jb], wa);
        if ((unsigned)(jb + 1) < (unsigned)NUM_STEPS)
          atomicAdd(&myH[base + jb + 1], wb);
      }
    }
  }
  __syncthreads();

  // Cross-wave reduce then one global float atomic per touched bin.
  for (int b = tid; b < HIST2; b += 256) {
    float sum = 0.0f;
#pragma unroll
    for (int w = 0; w < WAVES_PER_BLOCK; ++w) sum += sH[w * HIST2 + b];
    if (sum != 0.0f) unsafeAtomicAdd(&gHist[b], sum);
  }
}

// --- kernel 3: normalize + cumsum + dot -> scalar loss -----------------------
__global__ void finalize_kernel(const int* __restrict__ classes,
                                const float* __restrict__ gHist,
                                float* __restrict__ out, int n) {
  __shared__ int cc[NUM_CLASSES];
  const int tid = threadIdx.x;
  for (int c = tid; c < NUM_CLASSES; c += blockDim.x) cc[c] = 0;
  __syncthreads();
  for (int i = tid; i < n; i += blockDim.x) atomicAdd(&cc[classes[i]], 1);
  __syncthreads();
  if (tid == 0) {
    long long pos = 0;
    for (int c = 0; c < NUM_CLASSES; ++c) {
      long long m = cc[c];
      pos += m * (m - 1) / 2;
    }
    const long long tot = (long long)n * (n - 1) / 2;
    const long long neg = tot - pos;
    const float ip = pos > 0 ? 1.0f / (float)pos : 0.0f;
    const float in_ = neg > 0 ? 1.0f / (float)neg : 0.0f;
    float cdf = 0.0f, loss = 0.0f;
    for (int k = 0; k < NUM_STEPS; ++k) {
      cdf += gHist[k] * ip;                 // cumsum(hist_pos)
      loss += gHist[HIST_PAD + k] * in_ * cdf;
    }
    out[0] = loss;
  }
}

extern "C" void kernel_launch(void* const* d_in, const int* in_sizes, int n_in,
                              void* d_out, int out_size, void* d_ws,
                              size_t ws_size, hipStream_t stream) {
  const float* feats = (const float*)d_in[0];
  const int* classes = (const int*)d_in[1];
  float* out = (float*)d_out;

  const int n = in_sizes[1];            // 4096
  const int d = in_sizes[0] / n;        // 512

  float* gHist = (float*)d_ws;                              // 304 floats
  _Float16* Fh = (_Float16*)((char*)d_ws + 4096);           // n*d halfs

  zero_hist_kernel<<<1, HIST2, 0, stream>>>(gHist);

  const int nd = n * d;
  cvt_f16_kernel<<<(nd + 255) / 256, 256, 0, stream>>>(feats, Fh, nd);

  dim3 grid(n / TILE_N, n / TILE_M);    // (16, 32)
  hist_gemm_kernel<<<grid, 256, 0, stream>>>(Fh, classes, gHist, d);

  finalize_kernel<<<1, 256, 0, stream>>>(classes, gHist, out, n);
}
